// Model_19456201851192
// MI455X (gfx1250) — matrix operand order repaired and checked by reading, not compiled, not run on hardware
//
#include <hip/hip_runtime.h>
#include <math.h>

#define NN 8192
#define NE 49152
#define NK 32768
#define NL 128
#define NS 15
#define RAD 0.03f
#define LNEPS 1e-5f

typedef __attribute__((ext_vector_type(16))) __bf16 v16bf;
typedef __attribute__((ext_vector_type(2)))  __bf16 v2bf;
typedef __attribute__((ext_vector_type(8)))  float  v8f;

#if defined(__has_builtin)
#  if __has_builtin(__builtin_amdgcn_cvt_pk_bf16_f32)
#    define HAVE_PK_BF16 1
#  endif
#endif
#ifndef HAVE_PK_BF16
#  define HAVE_PK_BF16 0
#endif

__device__ __forceinline__ unsigned short f2bfu(float f) {
  union { float f; unsigned u; } v; v.f = f;
  unsigned r = v.u + 0x7fffu + ((v.u >> 16) & 1u);
  return (unsigned short)(r >> 16);
}

// Pack 16 f32 -> v16bf (RNE). Element pair (2j,2j+1) = VGPR j lo/hi halves.
__device__ __forceinline__ v16bf pack16(const float* f) {
#if HAVE_PK_BF16
  union { v16bf v; v2bf p[8]; } a;
#pragma unroll
  for (int j = 0; j < 8; ++j) a.p[j] = __builtin_amdgcn_cvt_pk_bf16_f32(f[2 * j], f[2 * j + 1]);
  return a.v;
#else
  union { v16bf v; unsigned short s[16]; } a;
#pragma unroll
  for (int j = 0; j < 16; ++j) a.s[j] = f2bfu(f[j]);
  return a.v;
#endif
}

// ---------------- weight conversion: f32 [cnt][K][N] -> bf16 [cnt][Kp][Np] (zero padded)
__global__ void cvt_kernel(const float* __restrict__ src, unsigned short* __restrict__ dst,
                           int Kd, int Nd, int Kp, int Np, int total) {
  int i = blockIdx.x * 256 + threadIdx.x;
  if (i >= total) return;
  int per = Kp * Np;
  int m = i / per, rem = i % per;
  int k = rem / Np, n = rem % Np;
  float v = (k < Kd && n < Nd) ? src[((size_t)m * Kd + k) * Nd + n] : 0.0f;
  dst[i] = f2bfu(v);
}

// ---------------- mesh-edge bitmask (N x N bits)
__global__ void mesh_mask_kernel(const int* __restrict__ s, const int* __restrict__ r,
                                 int e, unsigned* __restrict__ mask) {
  int i = blockIdx.x * 256 + threadIdx.x;
  if (i >= e) return;
  unsigned long long bit = ((unsigned long long)s[i] << 13) | (unsigned)r[i];
  atomicOr(&mask[bit >> 5], 1u << (bit & 31));
}

__device__ __forceinline__ bool radius_conn(float px, float py, float pz, float sqi,
                                            const float* wp, int i, int j,
                                            const unsigned* mask) {
  if (j == i) return false;
  float qx = wp[j * 3], qy = wp[j * 3 + 1], qz = wp[j * 3 + 2];
  float sqj = qx * qx + qy * qy + qz * qz;
  float dot = px * qx + py * qy + pz * qz;
  float d2 = sqi + sqj - 2.0f * dot;
  float dist = sqrtf(fmaxf(d2, 0.0f));
  if (!(dist < RAD)) return false;
  unsigned long long bit = ((unsigned long long)i << 13) | (unsigned)j;
  return !((mask[bit >> 5] >> (bit & 31)) & 1u);
}

__global__ __launch_bounds__(256) void radius_count_kernel(const float* __restrict__ wp,
                                                           const unsigned* __restrict__ mask,
                                                           int* __restrict__ rowcnt) {
  int i = blockIdx.x;
  float px = wp[i * 3], py = wp[i * 3 + 1], pz = wp[i * 3 + 2];
  float sqi = px * px + py * py + pz * pz;
  __shared__ int cnt;
  if (threadIdx.x == 0) cnt = 0;
  __syncthreads();
  int local = 0;
  for (int j = threadIdx.x; j < NN; j += 256)
    if (radius_conn(px, py, pz, sqi, wp, i, j, mask)) local++;
  atomicAdd(&cnt, local);
  __syncthreads();
  if (threadIdx.x == 0) rowcnt[i] = cnt;
}

__global__ void radius_fill_kernel(const float* __restrict__ wp, const unsigned* __restrict__ mask,
                                   const int* __restrict__ rowoff, int* __restrict__ ws,
                                   int* __restrict__ wr, int* __restrict__ valid) {
  int i = blockIdx.x * 64 + threadIdx.x;
  if (i >= NN) return;
  float px = wp[i * 3], py = wp[i * 3 + 1], pz = wp[i * 3 + 2];
  float sqi = px * px + py * py + pz * pz;
  int pos = rowoff[i];
  for (int j = 0; j < NN; ++j) {
    if (radius_conn(px, py, pz, sqi, wp, i, j, mask)) {
      if (pos < NK) { ws[pos] = i; wr[pos] = j; valid[pos] = 1; }
      pos++;
    }
  }
}

// ---------------- exclusive scan, n = 8192, one block
__global__ __launch_bounds__(256) void scan_kernel(const int* __restrict__ in, int* __restrict__ out, int n) {
  __shared__ int part[256];
  int t = threadIdx.x;
  int chunk = (n + 255) / 256;
  int base = t * chunk;
  int s = 0;
  for (int i = 0; i < chunk; ++i) { int p = base + i; if (p < n) s += in[p]; }
  part[t] = s;
  __syncthreads();
  if (t == 0) { int a = 0; for (int i = 0; i < 256; ++i) { int v = part[i]; part[i] = a; a += v; } }
  __syncthreads();
  int a = part[t];
  for (int i = 0; i < chunk; ++i) { int p = base + i; if (p < n) { out[p] = a; a += in[p]; } }
}

// ---------------- edge features
__global__ void mfeat_kernel(const float* __restrict__ mp, const float* __restrict__ wp,
                             const int* __restrict__ s, const int* __restrict__ r,
                             const float* __restrict__ mean, const float* __restrict__ stdv,
                             float* __restrict__ out, int e) {
  int i = blockIdx.x * 256 + threadIdx.x;
  if (i >= e) return;
  int a = s[i], b = r[i];
  float mx = mp[a*3] - mp[b*3], my = mp[a*3+1] - mp[b*3+1], mz = mp[a*3+2] - mp[b*3+2];
  float nm = sqrtf(mx*mx + my*my + mz*mz);
  float wx = wp[a*3] - wp[b*3], wy = wp[a*3+1] - wp[b*3+1], wz = wp[a*3+2] - wp[b*3+2];
  float nw = sqrtf(wx*wx + wy*wy + wz*wz);
  float f[8] = {mx, my, mz, nm, wx, wy, wz, nw};
  for (int c = 0; c < 8; ++c) out[(size_t)i * 8 + c] = (f[c] - mean[c]) / stdv[c];
}

__global__ void wfeat_kernel(const float* __restrict__ wp, const int* __restrict__ ws,
                             const int* __restrict__ wr, const float* __restrict__ mean,
                             const float* __restrict__ stdv, float* __restrict__ out, int k) {
  int i = blockIdx.x * 256 + threadIdx.x;
  if (i >= k) return;
  int a = ws[i], b = wr[i];
  float dx = wp[b*3] - wp[a*3], dy = wp[b*3+1] - wp[a*3+1], dz = wp[b*3+2] - wp[a*3+2];
  float nn = sqrtf(dx*dx + dy*dy + dz*dz);
  float f[4] = {dx, dy, dz, nn};
  for (int c = 0; c < 4; ++c) out[(size_t)i * 4 + c] = (f[c] - mean[c]) / stdv[c];
}

// ---------------- CSR build (deterministic via per-segment sort)
__global__ void csr_count_kernel(const int* __restrict__ recv, const int* __restrict__ valid,
                                 int n, int* __restrict__ cnt) {
  int e = blockIdx.x * 256 + threadIdx.x;
  if (e >= n) return;
  if (valid && !valid[e]) return;
  atomicAdd(&cnt[recv[e]], 1);
}

__global__ void csr_scatter_kernel(const int* __restrict__ recv, const int* __restrict__ valid,
                                   int n, const int* __restrict__ off, int* __restrict__ cur,
                                   int* __restrict__ csr) {
  int e = blockIdx.x * 256 + threadIdx.x;
  if (e >= n) return;
  if (valid && !valid[e]) return;
  int r = recv[e];
  int p = off[r] + atomicAdd(&cur[r], 1);
  csr[p] = e;
}

__global__ void csr_sort_kernel(const int* __restrict__ off, const int* __restrict__ cnt,
                                int* __restrict__ csr, int n) {
  int node = blockIdx.x * 128 + threadIdx.x;
  if (node >= n) return;
  int o = off[node], c = cnt[node];
  for (int i = 1; i < c; ++i) {
    int v = csr[o + i]; int j = i - 1;
    while (j >= 0 && csr[o + j] > v) { csr[o + j + 1] = csr[o + j]; --j; }
    csr[o + j + 1] = v;
  }
}

// ---------------- deterministic segment-sum gather
__global__ __launch_bounds__(128) void agg_kernel(const float* __restrict__ lat, const int* __restrict__ csr,
                                                  const int* __restrict__ off, const int* __restrict__ cnt,
                                                  float* __restrict__ agg) {
  int node = blockIdx.x, c = threadIdx.x;
  int o = off[node], k = cnt[node];
  float s = 0.0f;
  for (int i = 0; i < k; ++i) s += lat[(size_t)csr[o + i] * NL + c];
  agg[(size_t)node * NL + c] = s;
}

// ================= WMMA MLP machinery =================
// A fragment (16-bit A 16x32 layout): lane holds row (lane&15); lanes 0-15 carry
// K chunks [off..off+7] & [off+16..off+23]; lanes 16-31 the +8 chunks.
__device__ __forceinline__ v16bf afrag(const float* __restrict__ base, int off) {
  float f[16];
#pragma unroll
  for (int j = 0; j < 8; ++j) f[j] = base[off + j];
#pragma unroll
  for (int j = 0; j < 8; ++j) f[8 + j] = base[off + 16 + j];
  return pack16(f);
}

__device__ __forceinline__ v16bf afrag_guard(const float* __restrict__ base, int off, int fdim) {
  float f[16];
#pragma unroll
  for (int j = 0; j < 8; ++j) { int c = off + j;      f[j]     = (c < fdim) ? base[c] : 0.0f; }
#pragma unroll
  for (int j = 0; j < 8; ++j) { int c = off + 16 + j; f[8 + j] = (c < fdim) ? base[c] : 0.0f; }
  return pack16(f);
}

__device__ __forceinline__ v16bf afrag_onehot(int tp, int off) {
  float f[16];
#pragma unroll
  for (int j = 0; j < 8; ++j) { int c = off + j;      f[j]     = (c == tp) ? 1.0f : 0.0f; }
#pragma unroll
  for (int j = 0; j < 8; ++j) { int c = off + 16 + j; f[8 + j] = (c == tp) ? 1.0f : 0.0f; }
  return pack16(f);
}

// Two row-tiles (32 rows) share each B fragment: B lane L holds column N=t*16+(L&15),
// K chunks [kb+8h..+7] & [kb+16+8h..+7] (h = L>>4), matching the A fragment.
template <int NT>
__device__ __forceinline__ void wmma_block(const unsigned short* __restrict__ W, int wn, int kb,
                                           v16bf a0, v16bf a1, v8f* acc0, v8f* acc1, int lane) {
  const int bl15 = lane & 15, bhi = lane >> 4;
#pragma unroll
  for (int t = 0; t < NT; ++t) {
    union { v16bf v; unsigned short s[16]; } bu;
    const unsigned short* wc = W + t * 16 + bl15;
#pragma unroll
    for (int j = 0; j < 8; ++j) {
      bu.s[j]     = wc[(size_t)(kb + bhi * 8 + j) * wn];
      bu.s[8 + j] = wc[(size_t)(kb + 16 + bhi * 8 + j) * wn];
    }
    v16bf b = bu.v;
    acc0[t] = __builtin_amdgcn_wmma_f32_16x16x32_bf16(false, a0, false, b, (short)0, acc0[t], false, false);
    acc1[t] = __builtin_amdgcn_wmma_f32_16x16x32_bf16(false, a1, false, b, (short)0, acc1[t], false, false);
  }
}

template <int NT>
__device__ __forceinline__ void init_acc(v8f* acc, const float* __restrict__ bias, int dout, int l15) {
#pragma unroll
  for (int t = 0; t < NT; ++t) {
    int n = t * 16 + l15;
    float bv = (n < dout) ? bias[n] : 0.0f;
    v8f tmp = {bv, bv, bv, bv, bv, bv, bv, bv};
    acc[t] = tmp;
  }
}

template <int NT>
__device__ __forceinline__ void store_acc(const v8f* acc, bool relu, float* __restrict__ H, int hs,
                                          int rowoff, int l15, int hi) {
#pragma unroll
  for (int t = 0; t < NT; ++t) {
#pragma unroll
    for (int g = 0; g < 8; ++g) {
      float v = acc[t][g];
      if (relu) v = fmaxf(v, 0.0f);
      H[(rowoff + hi * 8 + g) * hs + t * 16 + l15] = v;  // D layout: lane group -> (M,N)
    }
  }
}

#define HSTRIDE 132

// 32-row tile x (128) @ bf16 W[128][wn] from f32 LDS in -> f32 LDS out
template <int NT>
__device__ __forceinline__ void gemm_lds(const float* __restrict__ H, const unsigned short* __restrict__ W,
                                         int wn, const float* __restrict__ bias, int dout, bool relu,
                                         float* __restrict__ O, int lane) {
  const int l15 = lane & 15, hi = lane >> 4;
  v8f acc0[NT], acc1[NT];
  init_acc<NT>(acc0, bias, dout, l15);
  init_acc<NT>(acc1, bias, dout, l15);
  const float* r0 = H + l15 * HSTRIDE;
  const float* r1 = H + (16 + l15) * HSTRIDE;
  for (int kb = 0; kb < 128; kb += 32) {
    int off = kb + hi * 8;
    v16bf a0 = afrag(r0, off);
    v16bf a1 = afrag(r1, off);
    wmma_block<NT>(W, wn, kb, a0, a1, acc0, acc1, lane);
  }
  store_acc<NT>(acc0, relu, O, HSTRIDE, 0, l15, hi);
  store_acc<NT>(acc1, relu, O, HSTRIDE, 16, l15, hi);
}

#define M_ONEHOT 0
#define M_DIRECT 1
#define M_GATHER3 2
#define M_PLAIN 3

__global__ __launch_bounds__(32) void mlp_kernel(
    int mode, int rows, int dinp,
    const float* __restrict__ src0, const int* __restrict__ idx0,
    const float* __restrict__ src1, const int* __restrict__ idx1,
    const float* __restrict__ src2, const int* __restrict__ idx2,
    const int* __restrict__ ntype, const float* __restrict__ feat, int fdim,
    const unsigned short* __restrict__ W0, const float* __restrict__ b0,
    const unsigned short* __restrict__ W1, const float* __restrict__ b1,
    const unsigned short* __restrict__ W2, const float* __restrict__ b2, int dout,
    const float* __restrict__ gam, const float* __restrict__ bet,
    const float* __restrict__ resid, float* __restrict__ out) {
  __shared__ float H1[32 * HSTRIDE];
  __shared__ float H2[32 * HSTRIDE];
  const int lane = threadIdx.x;
  const int l15 = lane & 15, hi = lane >> 4;
  const int row0 = blockIdx.x * 32;
  if (row0 >= rows) return;
  const int rA = row0 + l15, rB = row0 + 16 + l15;

  // ---- layer 1: A fragments built directly from global (no LDS staging).
  // Concat segments are 128-aligned; each 8-wide chunk stays inside one segment.
  v8f acc0[8], acc1[8];
  init_acc<8>(acc0, b0, 128, l15);
  init_acc<8>(acc1, b0, 128, l15);

  if (mode == M_GATHER3) {
    const float* pA[3]; const float* pB[3];
    pA[0] = src0 + (size_t)(idx0 ? idx0[rA] : rA) * NL;
    pB[0] = src0 + (size_t)(idx0 ? idx0[rB] : rB) * NL;
    pA[1] = src1 + (size_t)(idx1 ? idx1[rA] : rA) * NL;
    pB[1] = src1 + (size_t)(idx1 ? idx1[rB] : rB) * NL;
    pA[2] = src2 + (size_t)(idx2 ? idx2[rA] : rA) * NL;
    pB[2] = src2 + (size_t)(idx2 ? idx2[rB] : rB) * NL;
#pragma unroll
    for (int seg = 0; seg < 3; ++seg) {
      const float* qA = pA[seg];
      const float* qB = pB[seg];
#pragma unroll
      for (int kk = 0; kk < 128; kk += 32) {
        int off = kk + hi * 8;
        v16bf a0 = afrag(qA, off);
        v16bf a1 = afrag(qB, off);
        wmma_block<8>(W0, 128, seg * 128 + kk, a0, a1, acc0, acc1, lane);
      }
    }
  } else if (mode == M_PLAIN) {
    const float* qA = src0 + (size_t)rA * NL;
    const float* qB = src0 + (size_t)rB * NL;
    for (int kb = 0; kb < 128; kb += 32) {
      int off = kb + hi * 8;
      v16bf a0 = afrag(qA, off);
      v16bf a1 = afrag(qB, off);
      wmma_block<8>(W0, 128, kb, a0, a1, acc0, acc1, lane);
    }
  } else if (mode == M_DIRECT) {  // dinp == 32
    const float* qA = feat + (size_t)rA * fdim;
    const float* qB = feat + (size_t)rB * fdim;
    int off = hi * 8;
    v16bf a0 = afrag_guard(qA, off, fdim);
    v16bf a1 = afrag_guard(qB, off, fdim);
    wmma_block<8>(W0, 128, 0, a0, a1, acc0, acc1, lane);
  } else {  // M_ONEHOT, dinp == 32
    int tpa = ntype[rA], tpb = ntype[rB];
    int off = hi * 8;
    v16bf a0 = afrag_onehot(tpa, off);
    v16bf a1 = afrag_onehot(tpb, off);
    wmma_block<8>(W0, 128, 0, a0, a1, acc0, acc1, lane);
  }
  store_acc<8>(acc0, true, H1, HSTRIDE, 0, l15, hi);
  store_acc<8>(acc1, true, H1, HSTRIDE, 16, l15, hi);
  __syncthreads();

  // ---- layer 2
  gemm_lds<8>(H1, W1, 128, b1, 128, true, H2, lane);
  __syncthreads();

  // ---- layer 3 (+ LayerNorm/residual or 4-wide decoder write)
  if (dout == 128) {
    gemm_lds<8>(H2, W2, 128, b2, 128, false, H1, lane);
    __syncthreads();
    int row = row0 + lane;                 // 32 lanes, one row each
    const float* h = H1 + lane * HSTRIDE;
    float mu = 0.0f;
    for (int c = 0; c < NL; ++c) mu += h[c];
    mu *= (1.0f / 128.0f);
    float var = 0.0f;
    for (int c = 0; c < NL; ++c) { float d = h[c] - mu; var += d * d; }
    var *= (1.0f / 128.0f);
    float rstd = 1.0f / sqrtf(var + LNEPS);
    float* o = out + (size_t)row * NL;
    const float* rs = resid ? resid + (size_t)row * NL : nullptr;
    for (int c = 0; c < NL; ++c) {
      float v = (h[c] - mu) * rstd * gam[c] + bet[c];
      o[c] = rs ? rs[c] + v : v;
    }
  } else {
    gemm_lds<1>(H2, W2, 16, b2, dout, false, H1, lane);
    __syncthreads();
    for (int q = lane; q < 32 * dout; q += 32) {
      int r = q / dout, c = q % dout;
      out[(size_t)(row0 + r) * dout + c] = H1[r * HSTRIDE + c];
    }
  }
}

extern "C" void kernel_launch(void* const* d_in, const int* in_sizes, int n_in,
                              void* d_out, int out_size, void* d_ws, size_t ws_size,
                              hipStream_t stream) {
  const float* world_pos = (const float*)d_in[0];
  const float* mesh_pos  = (const float*)d_in[1];
  const int*   node_type = (const int*)d_in[2];
  const int*   senders   = (const int*)d_in[3];
  const int*   receivers = (const int*)d_in[4];
  const float* mesh_mean = (const float*)d_in[5];
  const float* mesh_std  = (const float*)d_in[6];
  const float* world_mean= (const float*)d_in[7];
  const float* world_std = (const float*)d_in[8];
  const float* enW0 = (const float*)d_in[9],  *enb0 = (const float*)d_in[10];
  const float* enW1 = (const float*)d_in[11], *enb1 = (const float*)d_in[12];
  const float* enW2 = (const float*)d_in[13], *enb2 = (const float*)d_in[14];
  const float* eng  = (const float*)d_in[15], *enbe = (const float*)d_in[16];
  const float* emW0 = (const float*)d_in[17], *emb0 = (const float*)d_in[18];
  const float* emW1 = (const float*)d_in[19], *emb1 = (const float*)d_in[20];
  const float* emW2 = (const float*)d_in[21], *emb2 = (const float*)d_in[22];
  const float* emg  = (const float*)d_in[23], *embe = (const float*)d_in[24];
  const float* ewW0 = (const float*)d_in[25], *ewb0 = (const float*)d_in[26];
  const float* ewW1 = (const float*)d_in[27], *ewb1 = (const float*)d_in[28];
  const float* ewW2 = (const float*)d_in[29], *ewb2 = (const float*)d_in[30];
  const float* ewg  = (const float*)d_in[31], *ewbe = (const float*)d_in[32];
  const float* peW0 = (const float*)d_in[33], *peb0 = (const float*)d_in[34];
  const float* peW1 = (const float*)d_in[35], *peb1 = (const float*)d_in[36];
  const float* peW2 = (const float*)d_in[37], *peb2 = (const float*)d_in[38];
  const float* peg  = (const float*)d_in[39], *pebe = (const float*)d_in[40];
  const float* pnW0 = (const float*)d_in[41], *pnb0 = (const float*)d_in[42];
  const float* pnW1 = (const float*)d_in[43], *pnb1 = (const float*)d_in[44];
  const float* pnW2 = (const float*)d_in[45], *pnb2 = (const float*)d_in[46];
  const float* png  = (const float*)d_in[47], *pnbe = (const float*)d_in[48];
  const float* dW0  = (const float*)d_in[49], *db0  = (const float*)d_in[50];
  const float* dW1  = (const float*)d_in[51], *db1  = (const float*)d_in[52];
  const float* dW2  = (const float*)d_in[53], *db2  = (const float*)d_in[54];

  size_t off = 0;
  auto alloc = [&](size_t bytes) -> void* {
    void* p = (char*)d_ws + off;
    off += (bytes + 255) & ~(size_t)255;
    return p;
  };
  float* node = (float*)alloc((size_t)NN * NL * 4);
  float* me   = (float*)alloc((size_t)NE * NL * 4);
  float* we   = (float*)alloc((size_t)NK * NL * 4);
  float* aggm = (float*)alloc((size_t)NN * NL * 4);
  float* aggw = (float*)alloc((size_t)NN * NL * 4);
  unsigned* mask = (unsigned*)alloc((size_t)NN * NN / 8);
  int* rowcnt = (int*)alloc(NN * 4);
  int* rowoff = (int*)alloc(NN * 4);
  int* ws_arr = (int*)alloc((size_t)NK * 4 * 3);
  int* wr_arr = ws_arr + NK;
  int* wvalid = wr_arr + NK;
  float* mfeat = (float*)alloc((size_t)NE * 8 * 4);
  float* wfeat = (float*)alloc((size_t)NK * 4 * 4);
  int* mcnt = (int*)alloc(NN * 4); int* moff = (int*)alloc(NN * 4);
  int* mcur = (int*)alloc(NN * 4); int* mcsr = (int*)alloc((size_t)NE * 4);
  int* wcnt = (int*)alloc(NN * 4); int* woff = (int*)alloc(NN * 4);
  int* wcur = (int*)alloc(NN * 4); int* wcsr = (int*)alloc((size_t)NK * 4);
  unsigned short* enW0b = (unsigned short*)alloc(32 * 128 * 2);
  unsigned short* enW1b = (unsigned short*)alloc(128 * 128 * 2);
  unsigned short* enW2b = (unsigned short*)alloc(128 * 128 * 2);
  unsigned short* emW0b = (unsigned short*)alloc(32 * 128 * 2);
  unsigned short* emW1b = (unsigned short*)alloc(128 * 128 * 2);
  unsigned short* emW2b = (unsigned short*)alloc(128 * 128 * 2);
  unsigned short* ewW0b = (unsigned short*)alloc(32 * 128 * 2);
  unsigned short* ewW1b = (unsigned short*)alloc(128 * 128 * 2);
  unsigned short* ewW2b = (unsigned short*)alloc(128 * 128 * 2);
  unsigned short* peW0b = (unsigned short*)alloc((size_t)30 * 384 * 128 * 2);
  unsigned short* peW1b = (unsigned short*)alloc((size_t)30 * 128 * 128 * 2);
  unsigned short* peW2b = (unsigned short*)alloc((size_t)30 * 128 * 128 * 2);
  unsigned short* pnW0b = (unsigned short*)alloc((size_t)15 * 384 * 128 * 2);
  unsigned short* pnW1b = (unsigned short*)alloc((size_t)15 * 128 * 128 * 2);
  unsigned short* pnW2b = (unsigned short*)alloc((size_t)15 * 128 * 128 * 2);
  unsigned short* dW0b = (unsigned short*)alloc(128 * 128 * 2);
  unsigned short* dW1b = (unsigned short*)alloc(128 * 128 * 2);
  unsigned short* dW2b = (unsigned short*)alloc(128 * 16 * 2);

  // ---- zero init
  hipMemsetAsync(mask, 0, (size_t)NN * NN / 8, stream);
  hipMemsetAsync(ws_arr, 0, (size_t)NK * 4 * 3, stream);
  hipMemsetAsync(mcnt, 0, NN * 4, stream);
  hipMemsetAsync(mcur, 0, NN * 4, stream);
  hipMemsetAsync(wcnt, 0, NN * 4, stream);
  hipMemsetAsync(wcur, 0, NN * 4, stream);

  // ---- weight conversion to padded bf16
  auto cvt = [&](const float* src, unsigned short* dst, int Kd, int Nd, int Kp, int Np, int cnt) {
    int total = cnt * Kp * Np;
    cvt_kernel<<<(total + 255) / 256, 256, 0, stream>>>(src, dst, Kd, Nd, Kp, Np, total);
  };
  cvt(enW0, enW0b, 9, 128, 32, 128, 1);   cvt(enW1, enW1b, 128, 128, 128, 128, 1); cvt(enW2, enW2b, 128, 128, 128, 128, 1);
  cvt(emW0, emW0b, 8, 128, 32, 128, 1);   cvt(emW1, emW1b, 128, 128, 128, 128, 1); cvt(emW2, emW2b, 128, 128, 128, 128, 1);
  cvt(ewW0, ewW0b, 4, 128, 32, 128, 1);   cvt(ewW1, ewW1b, 128, 128, 128, 128, 1); cvt(ewW2, ewW2b, 128, 128, 128, 128, 1);
  cvt(peW0, peW0b, 384, 128, 384, 128, 30); cvt(peW1, peW1b, 128, 128, 128, 128, 30); cvt(peW2, peW2b, 128, 128, 128, 128, 30);
  cvt(pnW0, pnW0b, 384, 128, 384, 128, 15); cvt(pnW1, pnW1b, 128, 128, 128, 128, 15); cvt(pnW2, pnW2b, 128, 128, 128, 128, 15);
  cvt(dW0, dW0b, 128, 128, 128, 128, 1);  cvt(dW1, dW1b, 128, 128, 128, 128, 1);   cvt(dW2, dW2b, 128, 4, 128, 16, 1);

  // ---- radius graph (row-major order, matching jnp.nonzero(size=K))
  mesh_mask_kernel<<<(NE + 255) / 256, 256, 0, stream>>>(senders, receivers, NE, mask);
  radius_count_kernel<<<NN, 256, 0, stream>>>(world_pos, mask, rowcnt);
  scan_kernel<<<1, 256, 0, stream>>>(rowcnt, rowoff, NN);
  radius_fill_kernel<<<NN / 64, 64, 0, stream>>>(world_pos, mask, rowoff, ws_arr, wr_arr, wvalid);

  // ---- edge features
  mfeat_kernel<<<(NE + 255) / 256, 256, 0, stream>>>(mesh_pos, world_pos, senders, receivers,
                                                     mesh_mean, mesh_std, mfeat, NE);
  wfeat_kernel<<<(NK + 255) / 256, 256, 0, stream>>>(world_pos, ws_arr, wr_arr,
                                                     world_mean, world_std, wfeat, NK);

  // ---- CSR for deterministic aggregation (mesh by receivers, world by wr & valid)
  csr_count_kernel<<<(NE + 255) / 256, 256, 0, stream>>>(receivers, nullptr, NE, mcnt);
  scan_kernel<<<1, 256, 0, stream>>>(mcnt, moff, NN);
  csr_scatter_kernel<<<(NE + 255) / 256, 256, 0, stream>>>(receivers, nullptr, NE, moff, mcur, mcsr);
  csr_sort_kernel<<<(NN + 127) / 128, 128, 0, stream>>>(moff, mcnt, mcsr, NN);
  csr_count_kernel<<<(NK + 255) / 256, 256, 0, stream>>>(wr_arr, wvalid, NK, wcnt);
  scan_kernel<<<1, 256, 0, stream>>>(wcnt, woff, NN);
  csr_scatter_kernel<<<(NK + 255) / 256, 256, 0, stream>>>(wr_arr, wvalid, NK, woff, wcur, wcsr);
  csr_sort_kernel<<<(NN + 127) / 128, 128, 0, stream>>>(woff, wcnt, wcsr, NN);

  // ---- encoders
  mlp_kernel<<<NN / 32, 32, 0, stream>>>(M_ONEHOT, NN, 32,
      nullptr, nullptr, nullptr, nullptr, nullptr, nullptr, node_type, nullptr, 0,
      enW0b, enb0, enW1b, enb1, enW2b, enb2, 128, eng, enbe, nullptr, node);
  mlp_kernel<<<NE / 32, 32, 0, stream>>>(M_DIRECT, NE, 32,
      nullptr, nullptr, nullptr, nullptr, nullptr, nullptr, nullptr, mfeat, 8,
      emW0b, emb0, emW1b, emb1, emW2b, emb2, 128, emg, embe, nullptr, me);
  mlp_kernel<<<NK / 32, 32, 0, stream>>>(M_DIRECT, NK, 32,
      nullptr, nullptr, nullptr, nullptr, nullptr, nullptr, nullptr, wfeat, 4,
      ewW0b, ewb0, ewW1b, ewb1, ewW2b, ewb2, 128, ewg, ewbe, nullptr, we);

  // ---- processor: S GraphNetBlocks
  for (int s = 0; s < NS; ++s) {
    size_t em_i = (size_t)(2 * s), ew_i = (size_t)(2 * s + 1);
    // mesh edge update (in-place residual)
    mlp_kernel<<<NE / 32, 32, 0, stream>>>(M_GATHER3, NE, 384,
        node, senders, node, receivers, me, nullptr, nullptr, nullptr, 0,
        peW0b + em_i * 384 * 128, peb0 + em_i * 128,
        peW1b + em_i * 128 * 128, peb1 + em_i * 128,
        peW2b + em_i * 128 * 128, peb2 + em_i * 128, 128,
        peg + em_i * 128, pebe + em_i * 128, me, me);
    // world edge update
    mlp_kernel<<<NK / 32, 32, 0, stream>>>(M_GATHER3, NK, 384,
        node, ws_arr, node, wr_arr, we, nullptr, nullptr, nullptr, 0,
        peW0b + ew_i * 384 * 128, peb0 + ew_i * 128,
        peW1b + ew_i * 128 * 128, peb1 + ew_i * 128,
        peW2b + ew_i * 128 * 128, peb2 + ew_i * 128, 128,
        peg + ew_i * 128, pebe + ew_i * 128, we, we);
    // aggregation (deterministic CSR gather)
    agg_kernel<<<NN, 128, 0, stream>>>(me, mcsr, moff, mcnt, aggm);
    agg_kernel<<<NN, 128, 0, stream>>>(we, wcsr, woff, wcnt, aggw);
    // node update
    size_t sn = (size_t)s;
    mlp_kernel<<<NN / 32, 32, 0, stream>>>(M_GATHER3, NN, 384,
        node, nullptr, aggm, nullptr, aggw, nullptr, nullptr, nullptr, 0,
        pnW0b + sn * 384 * 128, pnb0 + sn * 128,
        pnW1b + sn * 128 * 128, pnb1 + sn * 128,
        pnW2b + sn * 128 * 128, pnb2 + sn * 128, 128,
        png + sn * 128, pnbe + sn * 128, node, node);
  }

  // ---- decoder (no LN, dout=4) -> d_out
  mlp_kernel<<<NN / 32, 32, 0, stream>>>(M_PLAIN, NN, 128,
      node, nullptr, nullptr, nullptr, nullptr, nullptr, nullptr, nullptr, 0,
      dW0b, db0, dW1b, db1, dW2b, db2, 4, nullptr, nullptr, nullptr, (float*)d_out);
}
